// ShiftGCNLayer_17703855194820
// MI455X (gfx1250) — matrix operand order repaired and checked by reading, not compiled, not run on hardware
//
#include <hip/hip_runtime.h>

// ---------------------------------------------------------------------------
// ShiftGCN layer fused as one bf16 WMMA GEMM + BN/ReLU epilogue (gfx1250).
//   out[b,o,t,k] = sum_{c,j} x[b,c,t,j] * Wbig[(o,k),(c,j)]
//   Wbig[(o,k),(c,j)] = W[o/16][o%16][c]*Apow[o/16][j][k] + (j==k)*Wr[o][c]
// GEMM: M = B*T = 19200, N = O*V = 3200, K = C*V = 1600 (196.6 GFLOPs bf16).
// x repacked to bf16 GEMM layout once; A/B tiles stream to LDS via
// global_load_async_to_lds_b128 with double buffering (ASYNCcnt pipeline).
// ---------------------------------------------------------------------------

typedef __bf16  v16bf __attribute__((ext_vector_type(16)));
typedef __bf16  v8bf  __attribute__((ext_vector_type(8)));
typedef float   v8f   __attribute__((ext_vector_type(8)));
typedef unsigned short u16;

#define BB   64
#define CC   64
#define TT   300
#define VV   25
#define OO   128
#define MTOT (BB * TT)     // 19200
#define NTOT (OO * VV)     // 3200
#define KTOT (CC * VV)     // 1600

__device__ __forceinline__ u16 f2bf(float f) {
    unsigned int u = __builtin_bit_cast(unsigned int, f);
    unsigned int r = u + 0x7FFFu + ((u >> 16) & 1u);   // round-to-nearest-even
    return (u16)(r >> 16);
}

// --- CDNA5 async copy: 16B global -> LDS, tracked by ASYNCcnt --------------
__device__ __forceinline__ void async_copy16(const void* g, void* lds_generic) {
    unsigned loff = (unsigned)(uintptr_t)lds_generic;            // low 32b = LDS offset
    unsigned long long ga = (unsigned long long)(uintptr_t)g;
    asm volatile("global_load_async_to_lds_b128 %0, %1, off"
                 :: "v"(loff), "v"(ga) : "memory");
}
__device__ __forceinline__ void wait_async0() {
    asm volatile("s_wait_asynccnt 0" ::: "memory");
}

// ---------------------------------------------------------------------------
// Kernel 1: A^1..A^8 (25x25 each), one block, LDS power iteration.
// ---------------------------------------------------------------------------
__global__ __launch_bounds__(256) void apow_kernel(const float* __restrict__ A,
                                                   float* __restrict__ apw) {
    __shared__ float Ab[VV * VV];
    __shared__ float cur[VV * VV];
    __shared__ float nxt[VV * VV];
    const int tid = threadIdx.x;
    for (int i = tid; i < VV * VV; i += 256) { Ab[i] = A[i]; cur[i] = A[i]; }
    __syncthreads();
    for (int i = tid; i < VV * VV; i += 256) apw[i] = cur[i];
    for (int p = 1; p < 8; ++p) {
        for (int i = tid; i < VV * VV; i += 256) {
            int j = i / VV, k = i - j * VV;
            float s = 0.f;
            #pragma unroll
            for (int l = 0; l < VV; ++l) s += cur[j * VV + l] * Ab[l * VV + k];
            nxt[i] = s;
        }
        __syncthreads();
        for (int i = tid; i < VV * VV; i += 256) {
            cur[i] = nxt[i];
            apw[p * VV * VV + i] = nxt[i];
        }
        __syncthreads();
    }
}

// ---------------------------------------------------------------------------
// Kernel 2: combined weight, bf16, N-major [NTOT][KTOT].
// ---------------------------------------------------------------------------
__global__ __launch_bounds__(256) void wbig_kernel(const float* __restrict__ W,   // [128][64]
                                                   const float* __restrict__ Wr,  // [128][64]
                                                   const float* __restrict__ apw, // [8][25][25]
                                                   u16* __restrict__ wbigT) {
    int idx = blockIdx.x * 256 + threadIdx.x;
    if (idx >= NTOT * KTOT) return;
    int n  = idx / KTOT;
    int kk = idx - n * KTOT;
    int o  = n / VV,  kj = n - o * VV;
    int c  = kk / VV, j  = kk - c * VV;
    int br = o >> 4;
    float v = W[o * CC + c] * apw[(br * VV + j) * VV + kj];
    if (j == kj) v += Wr[o * CC + c];
    wbigT[idx] = f2bf(v);
}

// ---------------------------------------------------------------------------
// Kernel 3: repack x -> bf16 GEMM layout X2d[(b*300+t)][(c*25+j)].
// ---------------------------------------------------------------------------
__global__ __launch_bounds__(256) void repack_kernel(const float* __restrict__ x,
                                                     u16* __restrict__ xbf) {
    int idx = blockIdx.x * 256 + threadIdx.x;
    if (idx >= MTOT * KTOT) return;
    int m  = idx / KTOT;
    int kk = idx - m * KTOT;
    int b  = m / TT,  t = m - b * TT;
    int c  = kk / VV, j = kk - c * VV;
    xbf[idx] = f2bf(x[((size_t)(b * CC + c) * TT + t) * VV + j]);
}

// ---------------------------------------------------------------------------
// Kernel 4: bf16 WMMA GEMM, 256 threads = 8 waves (4M x 2N).
// Block tile 128x128, BK=32. Wave tile 32x64 -> 8 WMMAs / K-step.
// Async-to-LDS double-buffered staging.
// ---------------------------------------------------------------------------
#define LDK 40   // padded LDS row stride (elements); 80B spreads banks
__global__ __launch_bounds__(256) void gemm_kernel(const u16* __restrict__ xbf,
                                                   const u16* __restrict__ wbigT,
                                                   float* __restrict__ out) {
    __shared__ __attribute__((aligned(16))) __bf16 As[2][128][LDK];
    __shared__ __attribute__((aligned(16))) __bf16 Bs[2][128][LDK];

    const int tid  = threadIdx.x;
    const int lane = tid & 31;
    const int w    = tid >> 5;
    const int wm   = w >> 1;      // 0..3 : 32-row M strip
    const int wn   = w & 1;       // 0..1 : 64-col N strip
    const int m0   = blockIdx.y * 128;
    const int n0   = blockIdx.x * 128;

    // staging: each thread owns one 16-element row-half of A and of B
    const int srow = tid >> 1;
    const int koff = (tid & 1) * 16;
    const u16* aptr = xbf   + (size_t)(m0 + srow) * KTOT + koff;
    const u16* bptr = wbigT + (size_t)(n0 + srow) * KTOT + koff;

    v8f acc[2][4] = {};

    // prologue: stage K-step 0 into buffer 0
    async_copy16(aptr, &As[0][srow][koff]);
    async_copy16(aptr + 8, &As[0][srow][koff + 8]);
    async_copy16(bptr, &Bs[0][srow][koff]);
    async_copy16(bptr + 8, &Bs[0][srow][koff + 8]);

    const int NSTEP = KTOT / 32;          // 50
    for (int it = 0; it < NSTEP; ++it) {
        const int cur = it & 1;
        wait_async0();
        __syncthreads();
        if (it + 1 < NSTEP) {             // prefetch next K-step into other buffer
            async_copy16(aptr + (it + 1) * 32, &As[cur ^ 1][srow][koff]);
            async_copy16(aptr + (it + 1) * 32 + 8, &As[cur ^ 1][srow][koff + 8]);
            async_copy16(bptr + (it + 1) * 32, &Bs[cur ^ 1][srow][koff]);
            async_copy16(bptr + (it + 1) * 32 + 8, &Bs[cur ^ 1][srow][koff + 8]);
        }

        // fragments per CDNA5 16-bit layout: lane=(idx,half h);
        // elems 0..7 = K[8h..8h+7], elems 8..15 = K[16+8h..16+8h+7]
        const int m = lane & 15;
        const int h = lane >> 4;
        v16bf a[2];
        #pragma unroll
        for (int r = 0; r < 2; ++r) {
            const __bf16* arow = &As[cur][32 * wm + 16 * r + m][0];
            v8bf lo = *reinterpret_cast<const v8bf*>(arow + 8 * h);
            v8bf hi = *reinterpret_cast<const v8bf*>(arow + 16 + 8 * h);
            a[r] = __builtin_shufflevector(lo, hi, 0,1,2,3,4,5,6,7,8,9,10,11,12,13,14,15);
        }
        #pragma unroll
        for (int s = 0; s < 4; ++s) {
            const __bf16* brow = &Bs[cur][64 * wn + 16 * s + m][0];
            v8bf lo = *reinterpret_cast<const v8bf*>(brow + 8 * h);
            v8bf hi = *reinterpret_cast<const v8bf*>(brow + 16 + 8 * h);
            v16bf bfr = __builtin_shufflevector(lo, hi, 0,1,2,3,4,5,6,7,8,9,10,11,12,13,14,15);
            acc[0][s] = __builtin_amdgcn_wmma_f32_16x16x32_bf16(
                false, a[0], false, bfr, (short)0, acc[0][s], false, false);
            acc[1][s] = __builtin_amdgcn_wmma_f32_16x16x32_bf16(
                false, a[1], false, bfr, (short)0, acc[1][s], false, false);
        }
        __syncthreads();
    }

    // scatter D into [B,O,T,V]
    const int m = lane & 15;
    const int h = lane >> 4;
    #pragma unroll
    for (int s = 0; s < 4; ++s) {
        int ng = n0 + 64 * wn + 16 * s + m;
        int o  = ng / VV;
        int kj = ng - o * VV;
        #pragma unroll
        for (int r = 0; r < 2; ++r) {
            #pragma unroll
            for (int rr = 0; rr < 8; ++rr) {
                int mg = m0 + 32 * wm + 16 * r + rr + 8 * h;
                int bb = mg / TT;
                int tt = mg - bb * TT;
                out[(((size_t)bb * OO + o) * TT + tt) * VV + kj] = acc[r][s][rr];
            }
        }
    }
}

// ---------------------------------------------------------------------------
// Kernel 5: per-channel batch stats -> scale/shift.
// ---------------------------------------------------------------------------
__global__ __launch_bounds__(256) void bnstats_kernel(const float* __restrict__ out,
                                                      const float* __restrict__ gamma,
                                                      const float* __restrict__ beta,
                                                      float* __restrict__ scale,
                                                      float* __restrict__ shift) {
    const int ch = blockIdx.x;
    const int tid = threadIdx.x;
    const float4* o4 = reinterpret_cast<const float4*>(out);
    const int Q = (TT * VV) / 4;          // 1875
    float s = 0.f, s2 = 0.f;
    for (int f = tid; f < BB * Q; f += 256) {
        int b = f / Q, q = f - b * Q;
        float4 v = o4[(size_t)(b * OO + ch) * Q + q];
        s  += v.x + v.y + v.z + v.w;
        s2 += v.x * v.x + v.y * v.y + v.z * v.z + v.w * v.w;
    }
    __shared__ float sh[256], sh2[256];
    sh[tid] = s; sh2[tid] = s2;
    __syncthreads();
    for (int st = 128; st > 0; st >>= 1) {
        if (tid < st) { sh[tid] += sh[tid + st]; sh2[tid] += sh2[tid + st]; }
        __syncthreads();
    }
    if (tid == 0) {
        const float N = (float)(BB * TT * VV);
        float mean = sh[0] / N;
        float var  = sh2[0] / N - mean * mean;
        float sc   = gamma[ch] * rsqrtf(var + 1e-5f);
        scale[ch]  = sc;
        shift[ch]  = beta[ch] - mean * sc;
    }
}

// ---------------------------------------------------------------------------
// Kernel 6: in-place affine BN + ReLU, float4.
// ---------------------------------------------------------------------------
__global__ __launch_bounds__(256) void bnapply_kernel(float* __restrict__ out,
                                                      const float* __restrict__ scale,
                                                      const float* __restrict__ shift) {
    const int i = blockIdx.x * 256 + threadIdx.x;
    const int total4 = (BB * OO * TT * VV) / 4;
    if (i >= total4) return;
    const int Q = (TT * VV) / 4;          // 1875
    int ch = (i / Q) % OO;
    float sc = scale[ch], sf = shift[ch];
    float4* o4 = reinterpret_cast<float4*>(out);
    float4 v = o4[i];
    v.x = fmaxf(v.x * sc + sf, 0.f);
    v.y = fmaxf(v.y * sc + sf, 0.f);
    v.z = fmaxf(v.z * sc + sf, 0.f);
    v.w = fmaxf(v.w * sc + sf, 0.f);
    o4[i] = v;
}

// ---------------------------------------------------------------------------
extern "C" void kernel_launch(void* const* d_in, const int* in_sizes, int n_in,
                              void* d_out, int out_size, void* d_ws, size_t ws_size,
                              hipStream_t stream) {
    const float* x     = (const float*)d_in[0];   // [64,64,300,25]
    const float* A     = (const float*)d_in[1];   // [25,25]
    const float* W     = (const float*)d_in[2];   // [8,16,64] == [128,64]
    const float* Wr    = (const float*)d_in[3];   // [128,64]
    // d_in[4] = br: zeros AND cancels exactly under BN mean subtraction
    const float* gamma = (const float*)d_in[5];   // [128]
    const float* beta  = (const float*)d_in[6];   // [128]
    float* out = (float*)d_out;

    // workspace layout (all 16B aligned)
    float* apw   = (float*)d_ws;                                    // 8*625 floats
    float* scale = apw + 5008;                                      // 128
    float* shift = scale + 128;                                     // 128
    u16* wbigT = (u16*)((char*)d_ws + 32768);                       // [3200][1600] bf16
    u16* xbf   = (u16*)((char*)d_ws + 32768 + (size_t)NTOT * KTOT * 2); // [19200][1600] bf16

    apow_kernel<<<1, 256, 0, stream>>>(A, apw);

    wbig_kernel<<<(NTOT * KTOT + 255) / 256, 256, 0, stream>>>(W, Wr, apw, wbigT);

    repack_kernel<<<(MTOT * KTOT + 255) / 256, 256, 0, stream>>>(x, xbf);

    dim3 grid(NTOT / 128, MTOT / 128);   // (25, 150)
    gemm_kernel<<<grid, 256, 0, stream>>>(xbf, wbigT, out);

    bnstats_kernel<<<OO, 256, 0, stream>>>(out, gamma, beta, scale, shift);

    bnapply_kernel<<<((BB * OO * TT * VV) / 4 + 255) / 256, 256, 0, stream>>>(out, scale, shift);
}
